// RWKV7Layer_51445118271948
// MI455X (gfx1250) — compile-verified, matrix operand order checked by reading, not run on hardware
//
#include <hip/hip_runtime.h>
#include <hip/hip_bf16.h>
#include <math.h>
#include <stdint.h>

// ---------------- problem constants (reference setup_inputs) ----------------
#define B_SZ   2
#define T_SEQ  2048
#define H_DIM  1024
#define NHEAD  16
#define HD     64
#define FF_DIM 4096
#define DW     64
#define DA     64
#define DG     128
#define W_SCALE_F (-0.6065306597126334f)

typedef __attribute__((ext_vector_type(16))) __bf16   v16bf;
typedef __attribute__((ext_vector_type(8)))  float    v8f;
typedef __attribute__((ext_vector_type(4)))  unsigned u32x4;
typedef __attribute__((ext_vector_type(8)))  int      i32x8;
typedef __attribute__((ext_vector_type(4)))  int      i32x4;

#if defined(__has_builtin)
#  if __has_builtin(__builtin_amdgcn_tensor_load_to_lds) && __has_builtin(__builtin_amdgcn_s_wait_tensorcnt)
#    define USE_TDM 1
#  endif
#endif
#ifndef USE_TDM
#  define USE_TDM 0
#endif

__device__ __forceinline__ unsigned short f2bf(float f) {
  unsigned u = __builtin_bit_cast(unsigned, f);
  u += 0x7FFFu + ((u >> 16) & 1u);            // round-to-nearest-even
  return (unsigned short)(u >> 16);
}
__device__ __forceinline__ float sigmoidf_(float x) { return 1.0f / (1.0f + expf(-x)); }

enum { EPI_NONE = 0, EPI_TANH, EPI_SIGMOID, EPI_SIG_BIAS, EPI_DECAY, EPI_RELU2, EPI_ADD_RES };

#if USE_TDM
// -------- Tensor Data Mover: async 2D bf16 tile (128 rows x 32 cols) -> LDS --------
// D# per CDNA5 ISA: group0 {count=1, lds_addr, global_addr, type=2};
// group1 {data_size=2B, pad_enable, pad_interval=16dw, pad_amount=1dw, dims, strides}.
// LDS pad (1 dword per 16) reproduces the 34-short padded row used by the frag reads.
__device__ __forceinline__ void tdm_load_tile_A(const unsigned short* gptr,
                                                unsigned lds_off, int K) {
  unsigned long long ga = (unsigned long long)(uintptr_t)gptr;
  u32x4 g0;
  g0[0] = 1u;                                            // count=1, is_restore=0
  g0[1] = lds_off;                                       // lds_addr [63:32]
  g0[2] = (unsigned)ga;                                  // global_addr [95:64]
  g0[3] = (unsigned)((ga >> 32) & 0x1FFFFFFu) | (2u << 30); // addr[56:32] | type=2
  i32x8 g1;
  const unsigned tile0 = 32, tile1 = 128;                // tile: 32 (K) x 128 (M)
  const unsigned td0 = (unsigned)K, td1 = tile1;         // tensor dims (no OOB use)
  const unsigned long long str0 = (unsigned long long)(unsigned)K; // row stride (elems)
  g1[0] = (int)((1u << 16) | (1u << 20) | (3u << 22));   // data_size=2B, pad_en, intv=16dw, amt=1dw
  g1[1] = (int)((td0 & 0xFFFFu) << 16);                  // tensor_dim0[15:0] @63:48
  g1[2] = (int)((td0 >> 16) | ((td1 & 0xFFFFu) << 16));  // dim0 hi | dim1 lo
  g1[3] = (int)((td1 >> 16) | (tile0 << 16));            // dim1 hi | tile_dim0
  g1[4] = (int)(tile1);                                  // tile_dim1 (tile_dim2=0)
  g1[5] = (int)(str0 & 0xFFFFFFFFu);                     // dim0_stride lo
  g1[6] = (int)((str0 >> 32) & 0xFFFFu);                 // dim0_stride hi
  g1[7] = 0;
  i32x4 gz = {0, 0, 0, 0};
#if __clang_major__ >= 23
  i32x8 gz8 = {0, 0, 0, 0, 0, 0, 0, 0};
  __builtin_amdgcn_tensor_load_to_lds(g0, g1, gz, gz, gz8, 0);
#else
  __builtin_amdgcn_tensor_load_to_lds(g0, g1, gz, gz, 0);
#endif
}
#endif

// =====================================================================
// bf16 WMMA GEMM:  C[M,N] = epi( A[M,K] @ Bw[K,N] ), A/B bf16, acc f32.
// block = 256 threads = 8 waves; block tile 128(M) x 64(N); K step 32.
// Wave grid 4x2; each wave computes a 2x2 grid of 16x16 WMMA tiles
// (4 x v_wmma_f32_16x16x32_bf16 per K chunk, fragments reused).
// =====================================================================
template<int EPI, int OBF>
__global__ __launch_bounds__(256)
void gemm_wmma_bf16(const unsigned short* __restrict__ A, const unsigned short* __restrict__ Bw,
                    const float* __restrict__ bias, const float* __restrict__ res,
                    void* __restrict__ Cv, int M, int N, int K)
{
  __shared__ unsigned short Alds[128][34];   // [m][k] bf16, 68B rows (pad: conflict-free)
  __shared__ unsigned short Blds[64][34];    // [n][k] bf16 (transposed on store)

  const int tid  = threadIdx.x;
  const int lane = tid & 31, wave = tid >> 5;
  const int half = lane >> 4, ln = lane & 15;
  const int mw = wave & 3, nw = wave >> 2;
  const int m0 = blockIdx.y * 128, n0 = blockIdx.x * 64;

  v8f acc00 = {}, acc01 = {}, acc10 = {}, acc11 = {};

  for (int k0 = 0; k0 < K; k0 += 32) {
#if USE_TDM
    if (wave == 0)
      tdm_load_tile_A(A + (size_t)m0 * K + k0, (unsigned)(uintptr_t)&Alds[0][0], K);
#else
    for (int i = tid; i < 512; i += 256) {               // 128x32 bf16 via b128 loads
      int rr = i >> 2, cb = (i & 3) * 8;
      u32x4 d = *(const u32x4*)(A + (size_t)(m0 + rr) * K + k0 + cb);
      unsigned* dst = (unsigned*)&Alds[rr][cb];
      dst[0] = d[0]; dst[1] = d[1]; dst[2] = d[2]; dst[3] = d[3];
    }
#endif
    for (int i = tid; i < 2048; i += 256) {              // B 32x64 -> LDS transposed
      int kr = i >> 6, nc = i & 63;
      Blds[nc][kr] = Bw[(size_t)(k0 + kr) * N + n0 + nc];
    }
    if (k0 + 32 < K)                                     // global_prefetch_b8 next B tile
      __builtin_prefetch(Bw + (size_t)(k0 + 32) * N + n0 + (tid & 63), 0, 1);
#if USE_TDM
    if (wave == 0) __builtin_amdgcn_s_wait_tensorcnt(0);
#endif
    __syncthreads();

    // fragment loads per ISA 16-bit A 16x32 layout (B mirrored: K-strip per column)
    union { unsigned u[8]; v16bf v; } fa0, fa1, fb0, fb1;
    const unsigned* ar0 = (const unsigned*)&Alds[mw * 32 + ln][0];
    const unsigned* ar1 = (const unsigned*)&Alds[mw * 32 + 16 + ln][0];
    const unsigned* br0 = (const unsigned*)&Blds[nw * 32 + ln][0];
    const unsigned* br1 = (const unsigned*)&Blds[nw * 32 + 16 + ln][0];
    #pragma unroll
    for (int i = 0; i < 8; ++i) {
      int kd = (i < 4) ? (i + 4 * half) : (4 + i + 4 * half);  // dword index of K-pair
      fa0.u[i] = ar0[kd]; fa1.u[i] = ar1[kd];
      fb0.u[i] = br0[kd]; fb1.u[i] = br1[kd];
    }
    acc00 = __builtin_amdgcn_wmma_f32_16x16x32_bf16(false, fa0.v, false, fb0.v, (short)0, acc00, false, false);
    acc01 = __builtin_amdgcn_wmma_f32_16x16x32_bf16(false, fa0.v, false, fb1.v, (short)0, acc01, false, false);
    acc10 = __builtin_amdgcn_wmma_f32_16x16x32_bf16(false, fa1.v, false, fb0.v, (short)0, acc10, false, false);
    acc11 = __builtin_amdgcn_wmma_f32_16x16x32_bf16(false, fa1.v, false, fb1.v, (short)0, acc11, false, false);
    __syncthreads();
  }

  #pragma unroll
  for (int ti = 0; ti < 2; ++ti) {
    #pragma unroll
    for (int tj = 0; tj < 2; ++tj) {
      v8f acc = (ti == 0) ? (tj == 0 ? acc00 : acc01) : (tj == 0 ? acc10 : acc11);
      const int n = n0 + nw * 32 + tj * 16 + ln;
      #pragma unroll
      for (int j = 0; j < 8; ++j) {
        int m = m0 + mw * 32 + ti * 16 + 8 * half + j;   // C/D VGPR-j row mapping
        float x = acc[j], y;
        if      (EPI == EPI_NONE)     y = x;
        else if (EPI == EPI_TANH)     y = tanhf(x);
        else if (EPI == EPI_SIGMOID)  y = sigmoidf_(x);
        else if (EPI == EPI_SIG_BIAS) y = sigmoidf_(x + bias[n]);
        else if (EPI == EPI_DECAY)    y = expf(W_SCALE_F * sigmoidf_(x + bias[n]));
        else if (EPI == EPI_RELU2)  { float t = fmaxf(x, 0.0f); y = t * t; }
        else                          y = x + res[(size_t)m * N + n];   // EPI_ADD_RES
        if (OBF) ((unsigned short*)Cv)[(size_t)m * N + n] = f2bf(y);
        else     ((float*)Cv)[(size_t)m * N + n] = y;
      }
    }
  }
}

// ---------------- fp32 -> bf16 tensor conversion (weights) ----------------
__global__ void cvt_bf16_kernel(const float* __restrict__ in, unsigned short* __restrict__ o, size_t n) {
  for (size_t i = (size_t)blockIdx.x * 256 + threadIdx.x; i < n; i += (size_t)gridDim.x * 256)
    o[i] = f2bf(in[i]);
}

// ---------------- reductions ----------------
__device__ __forceinline__ float blockReduce256(float v, float* red) {
  int tid = threadIdx.x;
  red[tid] = v; __syncthreads();
  #pragma unroll
  for (int s = 128; s > 0; s >>= 1) { if (tid < s) red[tid] += red[tid + s]; __syncthreads(); }
  float r = red[0]; __syncthreads();
  return r;
}
__device__ __forceinline__ float blockReduce64(float v, float* red) {
  int tid = threadIdx.x;
  red[tid] = v; __syncthreads();
  #pragma unroll
  for (int s = 32; s > 0; s >>= 1) { if (tid < s) red[tid] += red[tid + s]; __syncthreads(); }
  float r = red[0]; __syncthreads();
  return r;
}

// ---------------- LayerNorm (1 or 2 chained), one block per token ----------------
__global__ __launch_bounds__(256)
void ln_kernel(const float* __restrict__ x,
               const float* __restrict__ w1, const float* __restrict__ b1,
               const float* __restrict__ w2, const float* __restrict__ b2,
               float* __restrict__ hid, float* __restrict__ hs, int twoLN)
{
  __shared__ float red[256];
  const int tid = threadIdx.x;
  const size_t base = (size_t)blockIdx.x * H_DIM;
  float vals[4];
  float s = 0.f;
  #pragma unroll
  for (int i = 0; i < 4; ++i) { vals[i] = x[base + tid + i * 256]; s += vals[i]; }
  float mean = blockReduce256(s, red) * (1.0f / H_DIM);
  float s2 = 0.f;
  #pragma unroll
  for (int i = 0; i < 4; ++i) { float d = vals[i] - mean; s2 += d * d; }
  float rstd = rsqrtf(blockReduce256(s2, red) * (1.0f / H_DIM) + 1e-5f);
  #pragma unroll
  for (int i = 0; i < 4; ++i) {
    int c = tid + i * 256;
    vals[i] = (vals[i] - mean) * rstd * w1[c] + b1[c];
    if (hid) hid[base + c] = vals[i];
  }
  if (twoLN) {
    s = 0.f;
    #pragma unroll
    for (int i = 0; i < 4; ++i) s += vals[i];
    mean = blockReduce256(s, red) * (1.0f / H_DIM);
    s2 = 0.f;
    #pragma unroll
    for (int i = 0; i < 4; ++i) { float d = vals[i] - mean; s2 += d * d; }
    rstd = rsqrtf(blockReduce256(s2, red) * (1.0f / H_DIM) + 1e-5f);
    #pragma unroll
    for (int i = 0; i < 4; ++i) {
      int c = tid + i * 256;
      vals[i] = (vals[i] - mean) * rstd * w2[c] + b2[c];
    }
  }
  #pragma unroll
  for (int i = 0; i < 4; ++i) hs[base + tid + i * 256] = vals[i];
}

// ---------------- token-shift mixes (write bf16 GEMM operands) ----------------
__global__ void mix6_kernel(const float* __restrict__ hs,
    const float* __restrict__ cr, const float* __restrict__ cw, const float* __restrict__ ck,
    const float* __restrict__ cv, const float* __restrict__ ca, const float* __restrict__ cg,
    unsigned short* __restrict__ xr, unsigned short* __restrict__ xw, unsigned short* __restrict__ xk,
    unsigned short* __restrict__ xv, unsigned short* __restrict__ xa, unsigned short* __restrict__ xg,
    size_t total)
{
  for (size_t idx = (size_t)blockIdx.x * 256 + threadIdx.x; idx < total;
       idx += (size_t)gridDim.x * 256) {
    int col = (int)(idx & (H_DIM - 1));
    int tin = (int)((idx >> 10) & (T_SEQ - 1));
    float cur = hs[idx];
    float prev = (tin == 0) ? 0.f : hs[idx - H_DIM];
    float d = prev - cur;
    xr[idx] = f2bf(cur + d * cr[col]); xw[idx] = f2bf(cur + d * cw[col]);
    xk[idx] = f2bf(cur + d * ck[col]); xv[idx] = f2bf(cur + d * cv[col]);
    xa[idx] = f2bf(cur + d * ca[col]); xg[idx] = f2bf(cur + d * cg[col]);
  }
}
__global__ void mix1_kernel(const float* __restrict__ hs, const float* __restrict__ ck,
                            unsigned short* __restrict__ xk, size_t total)
{
  for (size_t idx = (size_t)blockIdx.x * 256 + threadIdx.x; idx < total;
       idx += (size_t)gridDim.x * 256) {
    int col = (int)(idx & (H_DIM - 1));
    int tin = (int)((idx >> 10) & (T_SEQ - 1));
    float cur = hs[idx];
    float prev = (tin == 0) ? 0.f : hs[idx - H_DIM];
    xk[idx] = f2bf(cur + (prev - cur) * ck[col]);
  }
}

// ---------------- kk-normalize + k modification (per token, per head) ----------------
__global__ __launch_bounds__(64)
void prep_kernel(float* __restrict__ k_io, const float* __restrict__ a,
                 const float* __restrict__ k_k, const float* __restrict__ k_a,
                 float* __restrict__ avec, float* __restrict__ bvec)
{
  __shared__ float red[64];
  const int tid = threadIdx.x;
  const int token = blockIdx.x / NHEAD, h = blockIdx.x % NHEAD;
  const size_t idx = (size_t)token * H_DIM + h * HD + tid;
  const int hc = h * HD + tid;
  float kv  = k_io[idx];
  float kkh = kv * k_k[hc];
  float nrm = sqrtf(blockReduce64(kkh * kkh, red));
  float kkn = kkh / fmaxf(nrm, 1e-12f);
  float av  = a[idx];
  avec[idx] = -kkn;                                    // removal direction
  bvec[idx] = kkn * av;                                // replacement direction
  k_io[idx] = kv * (1.0f + (av - 1.0f) * k_a[hc]);     // in-place k update
}

// ---------------- delta-rule scan: 1 workgroup per (b,h); thread = value column ----------------
__global__ __launch_bounds__(64)
void recurrence_kernel(const float* __restrict__ r, const float* __restrict__ dec,
                       const float* __restrict__ kmod, const float* __restrict__ v,
                       const float* __restrict__ avec, const float* __restrict__ bvec,
                       float* __restrict__ o)
{
  __shared__ float r_s[64], d_s[64], k_s[64], a_s[64], b_s[64];
  const int tid = threadIdx.x;
  const int b = blockIdx.x / NHEAD, h = blockIdx.x % NHEAD;
  float S[HD];                                          // state column in VGPRs
  #pragma unroll
  for (int i = 0; i < HD; ++i) S[i] = 0.f;
  for (int t = 0; t < T_SEQ; ++t) {
    const size_t base = ((size_t)(b * T_SEQ + t)) * H_DIM + h * HD;
    r_s[tid] = r[base + tid];    d_s[tid] = dec[base + tid];
    k_s[tid] = kmod[base + tid]; a_s[tid] = avec[base + tid];
    b_s[tid] = bvec[base + tid];
    float vt = v[base + tid];
    __syncthreads();
    float sa = 0.f;
    #pragma unroll
    for (int kk = 0; kk < HD; ++kk) sa += a_s[kk] * S[kk];
    float ov = 0.f;
    #pragma unroll
    for (int kk = 0; kk < HD; ++kk) {
      S[kk] = S[kk] * d_s[kk] + b_s[kk] * sa + k_s[kk] * vt;
      ov += r_s[kk] * S[kk];
    }
    o[base + tid] = ov;
    __syncthreads();
  }
}

// ---------------- per-head groupnorm + bonus + output gate (bf16 out) ----------------
__global__ __launch_bounds__(64)
void gn_bonus_gate_kernel(const float* __restrict__ o, const float* __restrict__ r,
                          const float* __restrict__ kmod, const float* __restrict__ v,
                          const float* __restrict__ g, const float* __restrict__ r_k,
                          const float* __restrict__ gn_w, const float* __restrict__ gn_b,
                          unsigned short* __restrict__ og)
{
  __shared__ float red[64];
  const int tid = threadIdx.x;
  const int token = blockIdx.x / NHEAD, h = blockIdx.x % NHEAD;
  const size_t idx = (size_t)token * H_DIM + h * HD + tid;
  const int hc = h * HD + tid;
  float ov = o[idx];
  float mean = blockReduce64(ov, red) * (1.0f / HD);
  float d = ov - mean;
  float var = blockReduce64(d * d, red) * (1.0f / HD);
  float on = d * rsqrtf(var + (float)HD * 1e-5f);      // fla GroupNorm eps = HD*1e-5
  on = on * gn_w[hc] + gn_b[hc];
  float bsum = blockReduce64(r[idx] * kmod[idx] * r_k[hc], red);
  on += bsum * v[idx];
  og[idx] = f2bf(on * g[idx]);
}

// =====================================================================
extern "C" void kernel_launch(void* const* d_in, const int* in_sizes, int n_in,
                              void* d_out, int out_size, void* d_ws, size_t ws_size,
                              hipStream_t stream) {
  (void)in_sizes; (void)n_in; (void)out_size; (void)ws_size;
  const float* x      = (const float*)d_in[0];
  const float* pre_w  = (const float*)d_in[1];
  const float* pre_b  = (const float*)d_in[2];
  const float* attn_w = (const float*)d_in[3];
  const float* attn_b = (const float*)d_in[4];
  const float* ffn_w  = (const float*)d_in[5];
  const float* ffn_b  = (const float*)d_in[6];
  const float* x_r = (const float*)d_in[7];  const float* x_w = (const float*)d_in[8];
  const float* x_k = (const float*)d_in[9];  const float* x_v = (const float*)d_in[10];
  const float* x_a = (const float*)d_in[11]; const float* x_g = (const float*)d_in[12];
  const float* W_r = (const float*)d_in[13]; const float* W_k = (const float*)d_in[14];
  const float* W_v = (const float*)d_in[15]; const float* W_o = (const float*)d_in[16];
  const float* w1  = (const float*)d_in[17]; const float* w2  = (const float*)d_in[18];
  const float* w_b = (const float*)d_in[19];
  const float* a1  = (const float*)d_in[20]; const float* a2  = (const float*)d_in[21];
  const float* a_b = (const float*)d_in[22];
  const float* g1  = (const float*)d_in[23]; const float* g2  = (const float*)d_in[24];
  const float* k_k = (const float*)d_in[25]; const float* k_a = (const float*)d_in[26];
  const float* r_k = (const float*)d_in[27];
  const float* gn_w = (const float*)d_in[28]; const float* gn_b = (const float*)d_in[29];
  const float* ffn_x_k = (const float*)d_in[30];
  const float* W_key = (const float*)d_in[31]; const float* W_val = (const float*)d_in[32];
  float* out = (float*)d_out;

  const int MT = B_SZ * T_SEQ;                 // 4096 tokens
  const size_t MH = (size_t)MT * H_DIM;
  const size_t HH = (size_t)H_DIM * H_DIM;

  // ---- workspace carve-out ----
  float* ws = (float*)d_ws;
  size_t off = 0;
  auto allocF = [&](size_t n) { float* p = ws + off; off += n; return p; };
  auto allocH = [&](size_t n) { unsigned short* p = (unsigned short*)(ws + off); off += (n + 1) / 2; return p; };
  // fp32 intermediates
  float* hidden  = allocF(MH); float* hs = allocF(MH);
  float* rb = allocF(MH); float* kb = allocF(MH); float* vb = allocF(MH);
  float* decay = allocF(MH); float* ab = allocF(MH); float* gb = allocF(MH);
  float* avec = allocF(MH); float* bvec = allocF(MH);
  float* ob = allocF(MH); float* hidden2 = allocF(MH); float* hs2 = allocF(MH);
  // bf16 GEMM operands
  unsigned short* xr_b = allocH(MH); unsigned short* xw_b = allocH(MH);
  unsigned short* xk_b = allocH(MH); unsigned short* xv_b = allocH(MH);
  unsigned short* xa_b = allocH(MH); unsigned short* xg_b = allocH(MH);
  unsigned short* wt_b = allocH((size_t)MT * DW);
  unsigned short* at_b = allocH((size_t)MT * DA);
  unsigned short* gt_b = allocH((size_t)MT * DG);
  unsigned short* og_b = allocH(MH);
  unsigned short* xk2_b = allocH(MH);
  unsigned short* ff_b = allocH((size_t)MT * FF_DIM);
  // bf16 weights
  unsigned short* Wr_b = allocH(HH); unsigned short* Wk_b = allocH(HH);
  unsigned short* Wv_b = allocH(HH); unsigned short* Wo_b = allocH(HH);
  unsigned short* w1_b = allocH((size_t)H_DIM * DW); unsigned short* w2_b = allocH((size_t)DW * H_DIM);
  unsigned short* a1_b = allocH((size_t)H_DIM * DA); unsigned short* a2_b = allocH((size_t)DA * H_DIM);
  unsigned short* g1_b = allocH((size_t)H_DIM * DG); unsigned short* g2_b = allocH((size_t)DG * H_DIM);
  unsigned short* Wkey_b = allocH((size_t)H_DIM * FF_DIM);
  unsigned short* Wval_b = allocH((size_t)FF_DIM * H_DIM);

  const dim3 blk256(256), blk64(64);
  auto ggrid = [](int N, int M) { return dim3(N / 64, M / 128); };
  auto cgrid = [](size_t n) { size_t b = (n + 255) / 256; return dim3((unsigned)(b > 4096 ? 4096 : b)); };
  const int nEl = 8192;   // grid-stride elementwise over MH

  // 0) weight fp32 -> bf16 conversions
  cvt_bf16_kernel<<<cgrid(HH), blk256, 0, stream>>>(W_r, Wr_b, HH);
  cvt_bf16_kernel<<<cgrid(HH), blk256, 0, stream>>>(W_k, Wk_b, HH);
  cvt_bf16_kernel<<<cgrid(HH), blk256, 0, stream>>>(W_v, Wv_b, HH);
  cvt_bf16_kernel<<<cgrid(HH), blk256, 0, stream>>>(W_o, Wo_b, HH);
  cvt_bf16_kernel<<<cgrid(H_DIM*DW), blk256, 0, stream>>>(w1, w1_b, (size_t)H_DIM * DW);
  cvt_bf16_kernel<<<cgrid(H_DIM*DW), blk256, 0, stream>>>(w2, w2_b, (size_t)DW * H_DIM);
  cvt_bf16_kernel<<<cgrid(H_DIM*DA), blk256, 0, stream>>>(a1, a1_b, (size_t)H_DIM * DA);
  cvt_bf16_kernel<<<cgrid(H_DIM*DA), blk256, 0, stream>>>(a2, a2_b, (size_t)DA * H_DIM);
  cvt_bf16_kernel<<<cgrid(H_DIM*DG), blk256, 0, stream>>>(g1, g1_b, (size_t)H_DIM * DG);
  cvt_bf16_kernel<<<cgrid(H_DIM*DG), blk256, 0, stream>>>(g2, g2_b, (size_t)DG * H_DIM);
  cvt_bf16_kernel<<<cgrid((size_t)H_DIM*FF_DIM), blk256, 0, stream>>>(W_key, Wkey_b, (size_t)H_DIM * FF_DIM);
  cvt_bf16_kernel<<<cgrid((size_t)H_DIM*FF_DIM), blk256, 0, stream>>>(W_val, Wval_b, (size_t)FF_DIM * H_DIM);

  // 1) double LN: hidden = LN(x); hs = LN(hidden)
  ln_kernel<<<MT, blk256, 0, stream>>>(x, pre_w, pre_b, attn_w, attn_b, hidden, hs, 1);
  // 2) token-shift + six mixes (bf16 out)
  mix6_kernel<<<nEl, blk256, 0, stream>>>(hs, x_r, x_w, x_k, x_v, x_a, x_g,
                                          xr_b, xw_b, xk_b, xv_b, xa_b, xg_b, MH);
  // 3) projections + LoRAs (bf16 WMMA, fused activations)
  gemm_wmma_bf16<EPI_NONE,0><<<ggrid(H_DIM, MT), blk256, 0, stream>>>(xr_b, Wr_b, nullptr, nullptr, rb, MT, H_DIM, H_DIM);
  gemm_wmma_bf16<EPI_NONE,0><<<ggrid(H_DIM, MT), blk256, 0, stream>>>(xk_b, Wk_b, nullptr, nullptr, kb, MT, H_DIM, H_DIM);
  gemm_wmma_bf16<EPI_NONE,0><<<ggrid(H_DIM, MT), blk256, 0, stream>>>(xv_b, Wv_b, nullptr, nullptr, vb, MT, H_DIM, H_DIM);
  gemm_wmma_bf16<EPI_TANH,1><<<ggrid(DW, MT), blk256, 0, stream>>>(xw_b, w1_b, nullptr, nullptr, wt_b, MT, DW, H_DIM);
  gemm_wmma_bf16<EPI_DECAY,0><<<ggrid(H_DIM, MT), blk256, 0, stream>>>(wt_b, w2_b, w_b, nullptr, decay, MT, H_DIM, DW);
  gemm_wmma_bf16<EPI_NONE,1><<<ggrid(DA, MT), blk256, 0, stream>>>(xa_b, a1_b, nullptr, nullptr, at_b, MT, DA, H_DIM);
  gemm_wmma_bf16<EPI_SIG_BIAS,0><<<ggrid(H_DIM, MT), blk256, 0, stream>>>(at_b, a2_b, a_b, nullptr, ab, MT, H_DIM, DA);
  gemm_wmma_bf16<EPI_SIGMOID,1><<<ggrid(DG, MT), blk256, 0, stream>>>(xg_b, g1_b, nullptr, nullptr, gt_b, MT, DG, H_DIM);
  gemm_wmma_bf16<EPI_NONE,0><<<ggrid(H_DIM, MT), blk256, 0, stream>>>(gt_b, g2_b, nullptr, nullptr, gb, MT, H_DIM, DG);
  // 4) kk normalize, a/b vectors, k in-place update
  prep_kernel<<<MT * NHEAD, blk64, 0, stream>>>(kb, ab, k_k, k_a, avec, bvec);
  // 5) sequential delta-rule scan (32 independent chains)
  recurrence_kernel<<<B_SZ * NHEAD, blk64, 0, stream>>>(rb, decay, kb, vb, avec, bvec, ob);
  // 6) per-head groupnorm + bonus + gate (bf16 out)
  gn_bonus_gate_kernel<<<MT * NHEAD, blk64, 0, stream>>>(ob, rb, kb, vb, gb, r_k, gn_w, gn_b, og_b);
  // 7) output projection + residual
  gemm_wmma_bf16<EPI_ADD_RES,0><<<ggrid(H_DIM, MT), blk256, 0, stream>>>(og_b, Wo_b, nullptr, hidden, hidden2, MT, H_DIM, H_DIM);
  // 8) FFN: LN + token-shift mix (bf16)
  ln_kernel<<<MT, blk256, 0, stream>>>(hidden2, ffn_w, ffn_b, nullptr, nullptr, nullptr, hs2, 0);
  mix1_kernel<<<nEl, blk256, 0, stream>>>(hs2, ffn_x_k, xk2_b, MH);
  // 9) FFN GEMMs: relu^2 fused (bf16 out), then residual fused into d_out (f32)
  gemm_wmma_bf16<EPI_RELU2,1><<<ggrid(FF_DIM, MT), blk256, 0, stream>>>(xk2_b, Wkey_b, nullptr, nullptr, ff_b, MT, FF_DIM, H_DIM);
  gemm_wmma_bf16<EPI_ADD_RES,0><<<ggrid(H_DIM, MT), blk256, 0, stream>>>(ff_b, Wval_b, nullptr, hidden2, out, MT, H_DIM, FF_DIM);
}
